// Gemma4TextAttention_52046413693369
// MI455X (gfx1250) — matrix operand order repaired and checked by reading, not compiled, not run on hardware
//
#include <hip/hip_runtime.h>

// ---------------------------------------------------------------------------
// Gemma-style attention layer for gfx1250 (CDNA5), bf16 WMMA pipeline.
// B=2, S=2048, D=2048, H=16, KV=4, HD=256.
// Round 2: double-buffered software-pipelined GEMMs, batched fragment loads,
// 2x4 wave partition for better LDS-load : WMMA ratio.
// ---------------------------------------------------------------------------

#define H_    16
#define KV_   4
#define HD_   256
#define S_    2048
#define B_    2
#define D_    2048
#define EH_   (H_ * HD_)            // 4096
#define NQKV  (EH_ + 2 * KV_ * HD_) // 6144
#define M_    (B_ * S_)             // 4096

typedef __attribute__((ext_vector_type(16))) __bf16 v16bf;
typedef __attribute__((ext_vector_type(8)))  __bf16 v8bf;
typedef __attribute__((ext_vector_type(8)))  float  v8f;

// ---------------------------------------------------------------------------
// WMMA helpers
// ---------------------------------------------------------------------------
__device__ __forceinline__ v8f wmma_bf16(v16bf a, v16bf b, v8f c) {
  return __builtin_amdgcn_wmma_f32_16x16x32_bf16(false, a, false, b,
                                                 (short)0, c, false, false);
}

// A fragment: 16x32 bf16 (ISA layout).  p points at (row0, k0), ld in elems.
__device__ __forceinline__ v16bf load_frag_a(const __bf16* p, int ld) {
  const int lane = threadIdx.x & 31;
  const int row  = lane & 15;
  const int kb   = (lane >> 4) * 8;
  const __bf16* r = p + row * ld;
  v8bf lo = *(const v8bf*)(r + kb);
  v8bf hi = *(const v8bf*)(r + 16 + kb);
  v16bf out;
#pragma unroll
  for (int i = 0; i < 8; ++i) { out[i] = lo[i]; out[i + 8] = hi[i]; }
  return out;
}

// B fragment: 32x16 from N-major storage p[n*ld + k].
__device__ __forceinline__ v16bf load_frag_b(const __bf16* p, int ld) {
  const int lane = threadIdx.x & 31;
  const int n    = lane & 15;
  const int kb   = (lane >> 4) * 8;
  const __bf16* r = p + n * ld + kb;
  v8bf lo = *(const v8bf*)(r);
  v8bf hi = *(const v8bf*)(r + 16);
  v16bf out;
#pragma unroll
  for (int i = 0; i < 8; ++i) { out[i] = lo[i]; out[i + 8] = hi[i]; }
  return out;
}

__device__ __forceinline__ float rmax16(float v) {
  v = fmaxf(v, __shfl_xor(v, 1, 32));
  v = fmaxf(v, __shfl_xor(v, 2, 32));
  v = fmaxf(v, __shfl_xor(v, 4, 32));
  v = fmaxf(v, __shfl_xor(v, 8, 32));
  return v;
}
__device__ __forceinline__ float radd16(float v) {
  v += __shfl_xor(v, 1, 32);
  v += __shfl_xor(v, 2, 32);
  v += __shfl_xor(v, 4, 32);
  v += __shfl_xor(v, 8, 32);
  return v;
}

// ---------------------------------------------------------------------------
// Stage 0: conversion / layout kernels
// ---------------------------------------------------------------------------
__global__ __launch_bounds__(256) void cvt_hidden(const float* __restrict__ x,
                                                  __bf16* __restrict__ xb, int n) {
  int i = blockIdx.x * 256 + threadIdx.x;
  if (i < n) xb[i] = (__bf16)x[i];
}

__global__ __launch_bounds__(256) void cvt_wqkv_t(const float* __restrict__ Wq,
                                                  const float* __restrict__ Wk,
                                                  const float* __restrict__ Wv,
                                                  __bf16* __restrict__ Wt) {
  int k = blockIdx.x * 256 + threadIdx.x;
  int n = blockIdx.y;
  float v;
  if (n < EH_)                  v = Wq[(size_t)k * EH_ + n];
  else if (n < EH_ + KV_ * HD_) v = Wk[(size_t)k * (KV_ * HD_) + (n - EH_)];
  else                          v = Wv[(size_t)k * (KV_ * HD_) + (n - EH_ - KV_ * HD_)];
  Wt[(size_t)n * D_ + k] = (__bf16)v;
}

__global__ __launch_bounds__(256) void cvt_wo_t(const float* __restrict__ Wo,
                                                __bf16* __restrict__ WoT) {
  int k = blockIdx.x * 256 + threadIdx.x;
  int n = blockIdx.y;
  WoT[(size_t)n * EH_ + k] = (__bf16)Wo[(size_t)k * D_ + n];
}

// ---------------------------------------------------------------------------
// Stage 1: fused QKV GEMM + RMS-norm + RoPE epilogue.
// Grid: (M/64, 24).  64x256 tile per WG, waves 2(M) x 4(N).
// LDS: sX[2][64][40] | sW[2][256][40] bf16 | sAcc[64][264] f32
// ---------------------------------------------------------------------------
#define XBUF 2560   // 64*40 elems
#define WBUF 10240  // 256*40 elems
#define QKV_SMEM ((2 * XBUF + 2 * WBUF) * 2 + 64 * 264 * 4)

__global__ __launch_bounds__(256) void gemm_qkv_kernel(
    const __bf16* __restrict__ Xb, const __bf16* __restrict__ Wt,
    const float* __restrict__ qnw, const float* __restrict__ knw,
    const int* __restrict__ pos_ids,
    __bf16* __restrict__ qws, __bf16* __restrict__ kws, __bf16* __restrict__ vws) {
  extern __shared__ char smem[];
  __bf16* sX   = (__bf16*)smem;                         // [2][64][40]
  __bf16* sW   = sX + 2 * XBUF;                         // [2][256][40]
  float*  sAcc = (float*)(smem + (2 * XBUF + 2 * WBUF) * 2);

  const int m0   = blockIdx.x * 64;
  const int slab = blockIdx.y;
  const int n0   = slab * 256;
  const int tid  = threadIdx.x;
  const int wave = tid >> 5, lane = tid & 31;
  const int lh = lane >> 4, ln = lane & 15;
  const int wm = wave & 1, wn = wave >> 1;   // 2x4 waves over 64x256

  v8f acc[8] = {};

  const int xr = tid >> 2;
  const int xc = (tid & 3) * 8;
  const __bf16* xsrc = Xb + (size_t)(m0 + xr) * D_ + xc;
  const __bf16* wsrc = Wt + (size_t)(n0 + tid) * D_;

  // Prologue: tile 0 into registers
  v8bf xreg = *(const v8bf*)xsrc;
  v8bf wreg[4];
#pragma unroll
  for (int j = 0; j < 4; ++j) wreg[j] = *(const v8bf*)(wsrc + j * 8);

  const int STEPS = D_ / 32;
  for (int it = 0; it < STEPS; ++it) {
    __bf16* bX = sX + (it & 1) * XBUF;
    __bf16* bW = sW + (it & 1) * WBUF;
    *(v8bf*)&bX[xr * 40 + xc] = xreg;
#pragma unroll
    for (int j = 0; j < 4; ++j) *(v8bf*)&bW[tid * 40 + j * 8] = wreg[j];
    __syncthreads();

    if (it + 1 < STEPS) {  // next tile global->regs, overlapped with compute
      const int k0 = (it + 1) * 32;
      xreg = *(const v8bf*)(xsrc + k0);
#pragma unroll
      for (int j = 0; j < 4; ++j) wreg[j] = *(const v8bf*)(wsrc + k0 + j * 8);
      if (it + 2 < STEPS) __builtin_prefetch(wsrc + k0 + 32, 0, 1);
    }

    v16bf a0 = load_frag_a(bX + (wm * 32) * 40, 40);
    v16bf a1 = load_frag_a(bX + (wm * 32 + 16) * 40, 40);
    v16bf bb[4];
#pragma unroll
    for (int bi = 0; bi < 4; ++bi)
      bb[bi] = load_frag_b(bW + (wn * 64 + bi * 16) * 40, 40);
#pragma unroll
    for (int bi = 0; bi < 4; ++bi) acc[bi]     = wmma_bf16(a0, bb[bi], acc[bi]);
#pragma unroll
    for (int bi = 0; bi < 4; ++bi) acc[4 + bi] = wmma_bf16(a1, bb[bi], acc[4 + bi]);
  }

  // Spill accumulators: full 256-wide rows needed for norm/rope
#pragma unroll
  for (int ai = 0; ai < 2; ++ai)
#pragma unroll
    for (int bi = 0; bi < 4; ++bi)
#pragma unroll
      for (int i = 0; i < 8; ++i)
        sAcc[(wm * 32 + ai * 16 + i + 8 * lh) * 264 + wn * 64 + bi * 16 + ln] =
            acc[ai * 4 + bi][i];
  __syncthreads();

  // Epilogue: 4 threads per row, 64 cols each.
  const int r  = tid >> 2;
  const int cq = (tid & 3) * 64;
  const int m  = m0 + r;
  const int bb2 = m >> 11;
  const int s   = m & (S_ - 1);

  if (slab >= 20) {  // V: store transposed [B][KV][HD][S]
    const int g = slab - 20;
#pragma unroll 8
    for (int c0 = 0; c0 < 64; ++c0) {
      int c = cq + c0;
      vws[((size_t)(bb2 * KV_ + g) * HD_ + c) * S_ + s] = (__bf16)sAcc[r * 264 + c];
    }
  } else {           // Q / K: RMS-norm over HD, then RoPE
    const float* w = (slab < 16) ? qnw : knw;
    float ss = 0.f;
#pragma unroll 8
    for (int c0 = 0; c0 < 64; ++c0) {
      float v = sAcc[r * 264 + cq + c0];
      ss += v * v;
    }
    ss += __shfl_xor(ss, 1, 32);
    ss += __shfl_xor(ss, 2, 32);
    const float inv = rsqrtf(ss * (1.f / 256.f) + 1e-6f);
    const float pos = (float)pos_ids[s];
    const float kfreq = -0.07195664f;  // -ln(10000)/128
#pragma unroll 4
    for (int c0 = 0; c0 < 64; ++c0) {
      int c = cq + c0;
      int i = c & 127;
      float ang = pos * __expf(kfreq * (float)i);
      float sn, cs;
      __sincosf(ang, &sn, &cs);
      int pc = (c < 128) ? c + 128 : c - 128;
      float x  = sAcc[r * 264 + c]  * inv * (1.f + w[c]);
      float x2 = sAcc[r * 264 + pc] * inv * (1.f + w[pc]);
      float outv = (c < 128) ? (x * cs - x2 * sn) : (x * cs + x2 * sn);
      if (slab < 16)
        qws[((size_t)(bb2 * H_ + slab) * S_ + s) * HD_ + c] = (__bf16)outv;
      else
        kws[((size_t)(bb2 * KV_ + (slab - 16)) * S_ + s) * HD_ + c] = (__bf16)outv;
    }
  }
}

// ---------------------------------------------------------------------------
// Stage 2: causal flash attention.  Grid: (S/64 query tiles, B*H).
// Scores: waves 4(M)x2(N) (16q x 32k).  PV: waves 2(M)x4(N) (32q x 64hd).
// LDS: sQ[64][264] | sK[64][264] | sVt[256][72] | sP[64][72] | stats
// ---------------------------------------------------------------------------
#define ATTN_SMEM (2 * 64 * 264 * 2 + 256 * 72 * 2 + 64 * 72 * 2 + (3 * 64 + 2 * 128) * 4)

__global__ __launch_bounds__(256) void attn_kernel(
    const __bf16* __restrict__ qws, const __bf16* __restrict__ kws,
    const __bf16* __restrict__ vws, __bf16* __restrict__ aws) {
  extern __shared__ char smem[];
  __bf16* sQ  = (__bf16*)smem;         // [64][264]
  __bf16* sK  = sQ + 64 * 264;         // [64][264]
  __bf16* sVt = sK + 64 * 264;         // [256][72] (hd-major)
  __bf16* sP  = sVt + 256 * 72;        // [64][72]
  float* row_max = (float*)(sP + 64 * 72);
  float* row_sum = row_max + 64;
  float* row_scl = row_sum + 64;
  float* red_max = row_scl + 64;       // [64][2]
  float* red_sum = red_max + 128;      // [64][2]

  const int qt = blockIdx.x;
  const int bh = blockIdx.y;
  const int b  = bh >> 4, h = bh & 15, g = h >> 2;
  const int q0 = qt * 64;
  const int tid = threadIdx.x;
  const int wave = tid >> 5, lane = tid & 31;
  const int lh = lane >> 4, ln = lane & 15;
  const int wm = wave & 3, wn = wave >> 2;  // score partition: 4x2
  const int pm = wave & 1, pn = wave >> 1;  // PV partition:    2x4

  // Load Q tile (64x256 bf16)
  {
    int r = tid >> 2, seg = (tid & 3) * 64;
    const __bf16* src = qws + ((size_t)(b * H_ + h) * S_ + (q0 + r)) * HD_ + seg;
#pragma unroll
    for (int j = 0; j < 8; ++j)
      *(v8bf*)&sQ[r * 264 + seg + j * 8] = *(const v8bf*)(src + j * 8);
  }
  if (tid < 64) { row_max[tid] = -1e30f; row_sum[tid] = 0.f; }

  v8f o[8] = {};

  const __bf16* kbase = kws + (size_t)(b * KV_ + g) * S_ * HD_;
  const __bf16* vbase = vws + (size_t)(b * KV_ + g) * HD_ * S_;

  for (int ch = 0; ch <= qt; ++ch) {
    const int k0 = ch * 64;
    {
      int r = tid >> 2, seg = (tid & 3) * 64;
      const __bf16* src = kbase + (size_t)(k0 + r) * HD_ + seg;
#pragma unroll
      for (int j = 0; j < 8; ++j)
        *(v8bf*)&sK[r * 264 + seg + j * 8] = *(const v8bf*)(src + j * 8);
    }
    {
      const __bf16* src = vbase + (size_t)tid * S_ + k0;
#pragma unroll
      for (int j = 0; j < 8; ++j)
        *(v8bf*)&sVt[tid * 72 + j * 8] = *(const v8bf*)(src + j * 8);
    }
    if (ch < qt) {  // prefetch next chunk
      __builtin_prefetch(kbase + (size_t)(k0 + 64 + (tid >> 2)) * HD_ + (tid & 3) * 64, 0, 1);
      __builtin_prefetch(vbase + (size_t)tid * S_ + k0 + 64, 0, 1);
    }
    __syncthreads();

    // ---- scores: each wave 16q x 32keys, contraction HD=256
    v8f sc[2] = {};
#pragma unroll
    for (int kk = 0; kk < 8; ++kk) {
      v16bf a  = load_frag_a(sQ + wm * 16 * 264 + kk * 32, 264);
      v16bf b0 = load_frag_b(sK + (wn * 32) * 264 + kk * 32, 264);
      v16bf b1 = load_frag_b(sK + (wn * 32 + 16) * 264 + kk * 32, 264);
      sc[0] = wmma_bf16(a, b0, sc[0]);
      sc[1] = wmma_bf16(a, b1, sc[1]);
    }

    // ---- scale, causal mask, row max
    const float scale = 0.0625f;  // HD^-0.5
#pragma unroll
    for (int i = 0; i < 8; ++i) {
      const int rr = wm * 16 + i + 8 * lh;
      const int qrow = q0 + rr;
      float mx = -1e30f;
#pragma unroll
      for (int t = 0; t < 2; ++t) {
        float v = sc[t][i] * scale;
        int kcol = k0 + wn * 32 + t * 16 + ln;
        if (kcol > qrow) v = -1e30f;
        sc[t][i] = v;
        mx = fmaxf(mx, v);
      }
      mx = rmax16(mx);
      if (ln == 0) red_max[rr * 2 + wn] = mx;
    }
    __syncthreads();

    if (tid < 64) {
      float cm = fmaxf(red_max[tid * 2], red_max[tid * 2 + 1]);
      float mn = fmaxf(row_max[tid], cm);
      row_scl[tid] = __expf(row_max[tid] - mn);
      row_max[tid] = mn;
    }
    __syncthreads();

    // ---- P = exp(S - m): bf16 probs to LDS, row partial sums
#pragma unroll
    for (int i = 0; i < 8; ++i) {
      const int rr = wm * 16 + i + 8 * lh;
      const float mn = row_max[rr];
      float psum = 0.f;
#pragma unroll
      for (int t = 0; t < 2; ++t) {
        float p = __expf(sc[t][i] - mn);
        psum += p;
        sP[rr * 72 + wn * 32 + t * 16 + ln] = (__bf16)p;
      }
      psum = radd16(psum);
      if (ln == 0) red_sum[rr * 2 + wn] = psum;
    }
    __syncthreads();

    if (tid < 64)
      row_sum[tid] = row_sum[tid] * row_scl[tid] + red_sum[tid * 2] + red_sum[tid * 2 + 1];

    // ---- rescale running O (PV partition rows), then O += P * V
#pragma unroll
    for (int ai = 0; ai < 2; ++ai)
#pragma unroll
      for (int i = 0; i < 8; ++i) {
        const float scl = row_scl[pm * 32 + ai * 16 + i + 8 * lh];
#pragma unroll
        for (int bi = 0; bi < 4; ++bi) o[ai * 4 + bi][i] *= scl;
      }

#pragma unroll
    for (int kk = 0; kk < 2; ++kk) {
      v16bf a0 = load_frag_a(sP + (pm * 32) * 72 + kk * 32, 72);
      v16bf a1 = load_frag_a(sP + (pm * 32 + 16) * 72 + kk * 32, 72);
      v16bf bb[4];
#pragma unroll
      for (int bi = 0; bi < 4; ++bi)
        bb[bi] = load_frag_b(sVt + (pn * 64 + bi * 16) * 72 + kk * 32, 72);
#pragma unroll
      for (int bi = 0; bi < 4; ++bi) o[bi]     = wmma_bf16(a0, bb[bi], o[bi]);
#pragma unroll
      for (int bi = 0; bi < 4; ++bi) o[4 + bi] = wmma_bf16(a1, bb[bi], o[4 + bi]);
    }
    __syncthreads();
  }

  // ---- finalize: O /= l, store bf16 to attn workspace [B][S][H*HD]
#pragma unroll
  for (int ai = 0; ai < 2; ++ai)
#pragma unroll
    for (int i = 0; i < 8; ++i) {
      const int rr = pm * 32 + ai * 16 + i + 8 * lh;
      const float invl = 1.f / row_sum[rr];
#pragma unroll
      for (int bi = 0; bi < 4; ++bi) {
        const int c = pn * 64 + bi * 16 + ln;
        aws[(size_t)(b * S_ + q0 + rr) * EH_ + h * HD_ + c] =
            (__bf16)(o[ai * 4 + bi][i] * invl);
      }
    }
}

// ---------------------------------------------------------------------------
// Stage 3: output projection  out(f32) = attn(bf16 4096x4096) * Wo
// Grid: (M/64, D/256).  Double-buffered like stage 1.
// ---------------------------------------------------------------------------
#define OUT_SMEM ((2 * XBUF + 2 * WBUF) * 2)

__global__ __launch_bounds__(256) void gemm_out_kernel(
    const __bf16* __restrict__ Ab, const __bf16* __restrict__ WoT,
    float* __restrict__ out) {
  extern __shared__ char smem[];
  __bf16* sX = (__bf16*)smem;      // [2][64][40]
  __bf16* sW = sX + 2 * XBUF;      // [2][256][40]

  const int m0 = blockIdx.x * 64;
  const int n0 = blockIdx.y * 256;
  const int tid = threadIdx.x;
  const int wave = tid >> 5, lane = tid & 31;
  const int lh = lane >> 4, ln = lane & 15;
  const int wm = wave & 1, wn = wave >> 1;

  v8f acc[8] = {};
  const int xr = tid >> 2;
  const int xc = (tid & 3) * 8;
  const __bf16* xsrc = Ab + (size_t)(m0 + xr) * EH_ + xc;
  const __bf16* wsrc = WoT + (size_t)(n0 + tid) * EH_;

  v8bf xreg = *(const v8bf*)xsrc;
  v8bf wreg[4];
#pragma unroll
  for (int j = 0; j < 4; ++j) wreg[j] = *(const v8bf*)(wsrc + j * 8);

  const int STEPS = EH_ / 32;
  for (int it = 0; it < STEPS; ++it) {
    __bf16* bX = sX + (it & 1) * XBUF;
    __bf16* bW = sW + (it & 1) * WBUF;
    *(v8bf*)&bX[xr * 40 + xc] = xreg;
#pragma unroll
    for (int j = 0; j < 4; ++j) *(v8bf*)&bW[tid * 40 + j * 8] = wreg[j];
    __syncthreads();

    if (it + 1 < STEPS) {
      const int k0 = (it + 1) * 32;
      xreg = *(const v8bf*)(xsrc + k0);
#pragma unroll
      for (int j = 0; j < 4; ++j) wreg[j] = *(const v8bf*)(wsrc + k0 + j * 8);
      if (it + 2 < STEPS) __builtin_prefetch(wsrc + k0 + 32, 0, 1);
    }

    v16bf a0 = load_frag_a(bX + (wm * 32) * 40, 40);
    v16bf a1 = load_frag_a(bX + (wm * 32 + 16) * 40, 40);
    v16bf bb[4];
#pragma unroll
    for (int bi = 0; bi < 4; ++bi)
      bb[bi] = load_frag_b(bW + (wn * 64 + bi * 16) * 40, 40);
#pragma unroll
    for (int bi = 0; bi < 4; ++bi) acc[bi]     = wmma_bf16(a0, bb[bi], acc[bi]);
#pragma unroll
    for (int bi = 0; bi < 4; ++bi) acc[4 + bi] = wmma_bf16(a1, bb[bi], acc[4 + bi]);
  }

#pragma unroll
  for (int ai = 0; ai < 2; ++ai)
#pragma unroll
    for (int bi = 0; bi < 4; ++bi)
#pragma unroll
      for (int i = 0; i < 8; ++i)
        out[(size_t)(m0 + wm * 32 + ai * 16 + i + 8 * lh) * D_ +
            n0 + wn * 64 + bi * 16 + ln] = acc[ai * 4 + bi][i];
}

// ---------------------------------------------------------------------------
// Host launcher
// ---------------------------------------------------------------------------
extern "C" void kernel_launch(void* const* d_in, const int* in_sizes, int n_in,
                              void* d_out, int out_size, void* d_ws, size_t ws_size,
                              hipStream_t stream) {
  (void)in_sizes; (void)n_in; (void)out_size; (void)ws_size;
  const float* hs  = (const float*)d_in[0];
  const float* Wq  = (const float*)d_in[1];
  const float* Wk  = (const float*)d_in[2];
  const float* Wv  = (const float*)d_in[3];
  const float* Wo  = (const float*)d_in[4];
  const float* qnw = (const float*)d_in[5];
  const float* knw = (const float*)d_in[6];
  const int*   pid = (const int*)d_in[7];
  float* out = (float*)d_out;

  char* ws = (char*)d_ws;
  size_t o = 0;
  __bf16* Xb  = (__bf16*)(ws + o); o += (size_t)M_ * D_ * 2;
  __bf16* Wt  = (__bf16*)(ws + o); o += (size_t)NQKV * D_ * 2;
  __bf16* WoT = (__bf16*)(ws + o); o += (size_t)D_ * EH_ * 2;
  __bf16* qws = (__bf16*)(ws + o); o += (size_t)B_ * H_ * S_ * HD_ * 2;
  __bf16* kws = (__bf16*)(ws + o); o += (size_t)B_ * KV_ * S_ * HD_ * 2;
  __bf16* vws = (__bf16*)(ws + o); o += (size_t)B_ * KV_ * S_ * HD_ * 2;
  __bf16* aws = (__bf16*)(ws + o); o += (size_t)M_ * EH_ * 2;

  hipFuncSetAttribute(reinterpret_cast<const void*>(gemm_qkv_kernel),
                      hipFuncAttributeMaxDynamicSharedMemorySize, QKV_SMEM);
  hipFuncSetAttribute(reinterpret_cast<const void*>(attn_kernel),
                      hipFuncAttributeMaxDynamicSharedMemorySize, ATTN_SMEM);
  hipFuncSetAttribute(reinterpret_cast<const void*>(gemm_out_kernel),
                      hipFuncAttributeMaxDynamicSharedMemorySize, OUT_SMEM);

  cvt_hidden<<<(M_ * D_) / 256, 256, 0, stream>>>(hs, Xb, M_ * D_);
  cvt_wqkv_t<<<dim3(D_ / 256, NQKV), 256, 0, stream>>>(Wq, Wk, Wv, Wt);
  cvt_wo_t<<<dim3(EH_ / 256, D_), 256, 0, stream>>>(Wo, WoT);

  gemm_qkv_kernel<<<dim3(M_ / 64, NQKV / 256), 256, QKV_SMEM, stream>>>(
      Xb, Wt, qnw, knw, pid, qws, kws, vws);

  attn_kernel<<<dim3(S_ / 64, B_ * H_), 256, ATTN_SMEM, stream>>>(
      qws, kws, vws, aws);

  gemm_out_kernel<<<dim3(M_ / 64, D_ / 256), 256, OUT_SMEM, stream>>>(
      aws, WoT, out);
}
